// SelfAttention_70222715289780
// MI455X (gfx1250) — compile-verified, matrix operand order checked by reading, not run on hardware
//
#include <hip/hip_runtime.h>

typedef __attribute__((ext_vector_type(16))) _Float16 v16h;
typedef __attribute__((ext_vector_type(8)))  _Float16 v8h;
typedef __attribute__((ext_vector_type(8)))  float    v8f;
typedef __attribute__((ext_vector_type(4)))  unsigned int v4u;
typedef __attribute__((ext_vector_type(8)))  unsigned int v8u;

#define BATCH 8
#define SEQ   2048
#define DIM   512
#define NQKV  1536
#define MTOT  (BATCH * SEQ)      // 16384
#define QSCALE 0.044194173824159216f   // 1/sqrt(512)

#if __has_builtin(__builtin_amdgcn_s_wait_tensorcnt)
#define WAIT_TENSORCNT(n) __builtin_amdgcn_s_wait_tensorcnt(n)
#else
#define WAIT_TENSORCNT(n) asm volatile("s_wait_tensorcnt %0" ::"i"(n))
#endif

__device__ __forceinline__ v8f wmma16(v16h a, v16h b, v8f c) {
  return __builtin_amdgcn_wmma_f32_16x16x32_f16(false, a, false, b, (short)0, c,
                                                false, false);
}

// Load a 16x32 f16 A-fragment (or 32x16 B-fragment via its column pointer).
// CDNA5 layout: lane m = lane&15 holds K-chunks {0..7,16..23} (lanes 0-15)
// or {8..15,24..31} (lanes 16-31), contiguous along K.
__device__ __forceinline__ v16h load_frag(const _Float16* p, int stride, int lane) {
  int m    = lane & 15;
  int koff = (lane & 16) ? 8 : 0;
  const _Float16* q = p + m * stride + koff;
  v8h lo = *(const v8h*)q;
  v8h hi = *(const v8h*)(q + 16);
  v16h r;
#pragma unroll
  for (int i = 0; i < 8; ++i) { r[i] = lo[i]; r[i + 8] = hi[i]; }
  return r;
}

// ---- Tensor Data Mover: DMA a 128(rows)x32(cols) f16 tile from a row-major
// [rows_total x 512] tensor into LDS, padding each 64B row to 80B (40-half
// stride) via D# pad fields.  Issued wave-uniform; tracked by TENSORcnt.
__device__ __forceinline__ void tdm_load_tile_128x32(unsigned int lds_byte_addr,
                                                     const _Float16* tile0,
                                                     unsigned int rows_total) {
  unsigned long long ga = (unsigned long long)(size_t)tile0;
  v4u g0;
  g0[0] = 1u;                                      // count=1 (valid), user mode
  g0[1] = lds_byte_addr;                           // D#.lds_addr
  g0[2] = (unsigned int)ga;                        // global_addr[31:0]
  g0[3] = (unsigned int)((ga >> 32) & 0x01FFFFFFu) // global_addr[56:32]
        | (2u << 30);                              // type = 2 ("image")
  v8u g1;
  g1[0] = (1u << 16)                               // data_size = 2 bytes
        | (1u << 20)                               // pad_enable
        | (3u << 22)                               // pad_interval = 16 DWORDs
        | (3u << 25);                              // pad_amount   = 4 DWORDs
  g1[1] = ((unsigned int)DIM & 0xFFFFu) << 16;     // tensor_dim0 = 512
  g1[2] = (rows_total & 0xFFFFu) << 16;            // tensor_dim1[15:0]
  g1[3] = ((rows_total >> 16) & 0xFFFFu)           // tensor_dim1[31:16]
        | (32u << 16);                             // tile_dim0 = 32
  g1[4] = 128u;                                    // tile_dim1 = 128, tile_dim2 = 0
  g1[5] = (unsigned int)DIM;                       // tensor_dim0_stride = 512
  g1[6] = 0u;
  g1[7] = 0u;
  v4u gz; gz[0] = 0u; gz[1] = 0u; gz[2] = 0u; gz[3] = 0u;
  asm volatile("tensor_load_to_lds %0, %1, %2, %3"
               :: "s"(g0), "s"(g1), "s"(gz), "s"(gz)
               : "memory");
}

// ---------------- x: f32 -> f16 convert (once) ----------------
__global__ void convert_x_kernel(const float* __restrict__ x,
                                 _Float16* __restrict__ xh) {
  size_t i = ((size_t)blockIdx.x * 256 + threadIdx.x) * 8;
  if (i < (size_t)MTOT * DIM) {
    float4 a = *(const float4*)(x + i);
    float4 b = *(const float4*)(x + i + 4);
    v8h o;
    o[0] = (_Float16)a.x; o[1] = (_Float16)a.y;
    o[2] = (_Float16)a.z; o[3] = (_Float16)a.w;
    o[4] = (_Float16)b.x; o[5] = (_Float16)b.y;
    o[6] = (_Float16)b.z; o[7] = (_Float16)b.w;
    *(v8h*)(xh + i) = o;
  }
}

// ---------------- weight transpose + f32->f16 convert ----------------
__global__ void transpose_w_kernel(const float* __restrict__ w,
                                   _Float16* __restrict__ wt,
                                   int K, int Ncols) {
  int i = blockIdx.x * 256 + threadIdx.x;
  if (i < K * Ncols) {
    int k = i / Ncols, n = i % Ncols;
    wt[(size_t)n * K + k] = (_Float16)w[i];
  }
}

// ---------------- QKV GEMM: [16384,512] x [512,1536], TDM double-buffered ----------------
__global__ __launch_bounds__(256) void gemm_qkv_kernel(
    const _Float16* __restrict__ xh, const _Float16* __restrict__ wt,
    const float* __restrict__ bias, _Float16* __restrict__ qh,
    _Float16* __restrict__ kh, _Float16* __restrict__ vth) {
  __shared__ __align__(16) _Float16 As[2][128 * 40];
  __shared__ __align__(16) _Float16 Bs[2][128 * 40];
  const int tid = threadIdx.x, lane = tid & 31, wv = tid >> 5;
  const int m0 = blockIdx.y * 128, n0 = blockIdx.x * 128;
  const int wm = (wv & 3) * 32, wn = (wv >> 2) * 64;
  v8f acc[2][4] = {};

  const unsigned int asa0 = (unsigned int)(size_t)(void*)&As[0][0];
  const unsigned int asa1 = (unsigned int)(size_t)(void*)&As[1][0];
  const unsigned int bsa0 = (unsigned int)(size_t)(void*)&Bs[0][0];
  const unsigned int bsa1 = (unsigned int)(size_t)(void*)&Bs[1][0];
  const _Float16* abase = xh + (size_t)m0 * DIM;
  const _Float16* bbase = wt + (size_t)n0 * DIM;

  if (wv == 0) {                       // prologue: pair 0 into buffer 0
    tdm_load_tile_128x32(asa0, abase, MTOT);
    tdm_load_tile_128x32(bsa0, bbase, NQKV);
  }

  int step = 0;
  for (int k0 = 0; k0 < DIM; k0 += 32, ++step) {
    if (wv == 0) {
      if (k0 + 32 < DIM) {             // prefetch next pair into other buffer
        unsigned int na = (step & 1) ? asa0 : asa1;
        unsigned int nb = (step & 1) ? bsa0 : bsa1;
        tdm_load_tile_128x32(na, abase + k0 + 32, MTOT);
        tdm_load_tile_128x32(nb, bbase + k0 + 32, NQKV);
        WAIT_TENSORCNT(2);             // current pair has landed (in-order)
      } else {
        WAIT_TENSORCNT(0);
      }
    }
    __syncthreads();                   // all waves see current tiles
    const _Float16* a = As[step & 1];
    const _Float16* bb = Bs[step & 1];
    v16h a0 = load_frag(a + (wm + 0)  * 40, 40, lane);
    v16h a1 = load_frag(a + (wm + 16) * 40, 40, lane);
#pragma unroll
    for (int nt = 0; nt < 4; ++nt) {
      v16h bf = load_frag(bb + (wn + nt * 16) * 40, 40, lane);
      acc[0][nt] = wmma16(a0, bf, acc[0][nt]);
      acc[1][nt] = wmma16(a1, bf, acc[1][nt]);
    }
    __syncthreads();                   // done reading before buffer is refilled
  }

  const int part = n0 >> 9;   // 0=q, 1=k, 2=v (uniform per block)
#pragma unroll
  for (int nt = 0; nt < 4; ++nt) {
    int gc = n0 + wn + nt * 16 + (lane & 15);
    float bv = bias[gc];
#pragma unroll
    for (int t = 0; t < 2; ++t) {
#pragma unroll
      for (int r = 0; r < 8; ++r) {
        int gr = m0 + wm + t * 16 + r + ((lane >> 4) << 3);
        float v = acc[t][nt][r] + bv;
        if (part == 0) {
          qh[(size_t)gr * DIM + gc] = (_Float16)(v * QSCALE);
        } else if (part == 1) {
          kh[(size_t)gr * DIM + (gc - DIM)] = (_Float16)v;
        } else {
          int bb2 = gr >> 11, nr = gr & (SEQ - 1), d = gc - 2 * DIM;
          vth[(size_t)bb2 * DIM * SEQ + (size_t)d * SEQ + nr] = (_Float16)v;
        }
      }
    }
  }
}

// ---------------- Flash attention (TDM double-buffered K tiles) ----------------
// grid (64, 8), block 256.  32 Q rows per block; 8 waves: wave w owns
// S columns [16w,16w+16) and output D-slice [64w,64w+64).
__global__ __launch_bounds__(256) void attn_kernel(
    const _Float16* __restrict__ qh, const _Float16* __restrict__ kh,
    const _Float16* __restrict__ vth, _Float16* __restrict__ oh) {
  __shared__ __align__(16) _Float16 Qs[32 * 520];
  __shared__ __align__(16) _Float16 Ks[2][128 * 40];
  __shared__ __align__(16) _Float16 Ps[32 * 136];
  __shared__ float wmaxs[8 * 32];
  __shared__ float psums[8 * 32];
  __shared__ float row_m[32], row_l[32], row_sc[32];

  const int tid = threadIdx.x, lane = tid & 31, wv = tid >> 5;
  const int b = blockIdx.y;
  const int q0 = blockIdx.x * 32;

  const _Float16* qb = qh  + ((size_t)b * SEQ + q0) * DIM;
  const _Float16* kb = kh  + (size_t)b * SEQ * DIM;
  const _Float16* vb = vth + (size_t)b * DIM * SEQ;

  const unsigned int ksa0 = (unsigned int)(size_t)(void*)&Ks[0][0];
  const unsigned int ksa1 = (unsigned int)(size_t)(void*)&Ks[1][0];

  // prefetch first K tile (mc=0, d0=0) via TDM while Q is staged
  if (wv == 0) tdm_load_tile_128x32(ksa0, kb, SEQ);

  // stage full 32x512 Q block into LDS
  for (int idx = tid; idx < (32 * DIM) / 8; idx += 256) {
    int row = idx >> 6;
    int cc  = (idx & 63) * 8;
    *(v8h*)(Qs + row * 520 + cc) = *(const v8h*)(qb + (size_t)row * DIM + cc);
  }
  if (wv == 0) { row_m[lane] = -1e30f; row_l[lane] = 0.0f; }
  psums[tid] = 0.0f;

  v8f O[2][4] = {};
  int ti = 0;                      // flattened K-tile index (16 d0 x 16 mc)

  for (int mc = 0; mc < SEQ; mc += 128) {
    // ---- S = Q @ K^T for this 128-wide key chunk ----
    v8f S0 = {}, S1 = {};
    for (int d0 = 0; d0 < DIM; d0 += 32) {
      if (wv == 0) {
        int tn = ti + 1;
        if (tn < (SEQ / 128) * (DIM / 32)) {
          int nmc = (tn >> 4) * 128, nd0 = (tn & 15) * 32;
          tdm_load_tile_128x32((tn & 1) ? ksa1 : ksa0,
                               kb + (size_t)nmc * DIM + nd0, SEQ);
          WAIT_TENSORCNT(1);       // tile ti has landed (in-order TDM)
        } else {
          WAIT_TENSORCNT(0);
        }
      }
      __syncthreads();             // all waves see tile ti
      const _Float16* kbuf = Ks[ti & 1];
      v16h bf = load_frag(kbuf + (wv * 16) * 40, 40, lane);  // cols = K rows
      v16h a0 = load_frag(Qs + 0  * 520 + d0, 520, lane);
      v16h a1 = load_frag(Qs + 16 * 520 + d0, 520, lane);
      S0 = wmma16(a0, bf, S0);
      S1 = wmma16(a1, bf, S1);
      __syncthreads();             // done reading buf before it is refilled
      ++ti;
    }

    // ---- per-wave row max of its 16-col slice ----
#pragma unroll
    for (int t = 0; t < 2; ++t) {
      v8f s = t ? S1 : S0;
#pragma unroll
      for (int r = 0; r < 8; ++r) {
        float v = s[r];
        v = fmaxf(v, __shfl_xor(v, 1));
        v = fmaxf(v, __shfl_xor(v, 2));
        v = fmaxf(v, __shfl_xor(v, 4));
        v = fmaxf(v, __shfl_xor(v, 8));
        if ((lane & 15) == 0)
          wmaxs[wv * 32 + t * 16 + r + ((lane >> 4) << 3)] = v;
      }
    }
    __syncthreads();   // (A)

    // ---- softmax stats update (deterministic: fixed-order sums) ----
    if (wv == 0) {
      float mo = row_m[lane];
      float mx = mo;
#pragma unroll
      for (int j = 0; j < 8; ++j) mx = fmaxf(mx, wmaxs[j * 32 + lane]);
      float sc = __expf(mo - mx);
      float ps = 0.0f;
#pragma unroll
      for (int j = 0; j < 8; ++j) ps += psums[j * 32 + lane];
      row_sc[lane] = sc;
      row_m[lane]  = mx;
      row_l[lane]  = (row_l[lane] + ps) * sc;   // fold previous chunk's sums
    }
    __syncthreads();   // (B)

    // ---- P = exp(S - m); write P to LDS; per-wave row sums ----
#pragma unroll
    for (int t = 0; t < 2; ++t) {
      v8f s = t ? S1 : S0;
#pragma unroll
      for (int r = 0; r < 8; ++r) {
        int row = t * 16 + r + ((lane >> 4) << 3);
        float p = __expf(s[r] - row_m[row]);
        Ps[row * 136 + wv * 16 + (lane & 15)] = (_Float16)p;
        float v = p;
        v += __shfl_xor(v, 1);
        v += __shfl_xor(v, 2);
        v += __shfl_xor(v, 4);
        v += __shfl_xor(v, 8);
        if ((lane & 15) == 0) psums[wv * 32 + row] = v;
      }
    }
    // ---- rescale running output accumulator ----
#pragma unroll
    for (int t = 0; t < 2; ++t)
#pragma unroll
      for (int r = 0; r < 8; ++r) {
        float sc = row_sc[t * 16 + r + ((lane >> 4) << 3)];
#pragma unroll
        for (int nt = 0; nt < 4; ++nt) O[t][nt][r] *= sc;
      }
    __syncthreads();   // (C)

    // ---- O += P(32x128) @ V(128 x 64-slice), B-frags direct from V^T ----
#pragma unroll
    for (int kk = 0; kk < 128; kk += 32) {
      v16h a0 = load_frag(Ps + 0  * 136 + kk, 136, lane);
      v16h a1 = load_frag(Ps + 16 * 136 + kk, 136, lane);
#pragma unroll
      for (int nt = 0; nt < 4; ++nt) {
        const _Float16* vcol = vb + (size_t)(wv * 64 + nt * 16) * SEQ + mc + kk;
        v16h bf = load_frag(vcol, SEQ, lane);
        O[0][nt] = wmma16(a0, bf, O[0][nt]);
        O[1][nt] = wmma16(a1, bf, O[1][nt]);
      }
    }
  }
  __syncthreads();

  // ---- finalize: divide by row sums, store f16 attention output ----
  _Float16* ob = oh + ((size_t)b * SEQ + q0) * DIM;
#pragma unroll
  for (int t = 0; t < 2; ++t)
#pragma unroll
    for (int r = 0; r < 8; ++r) {
      int row = t * 16 + r + ((lane >> 4) << 3);
      float l = row_l[row];
#pragma unroll
      for (int j = 0; j < 8; ++j) l += psums[j * 32 + row];
      float inv = 1.0f / l;
#pragma unroll
      for (int nt = 0; nt < 4; ++nt) {
        int d = wv * 64 + nt * 16 + (lane & 15);
        ob[(size_t)row * DIM + d] = (_Float16)(O[t][nt][r] * inv);
      }
    }
}

// ---------------- FC GEMM + bias + residual, TDM double-buffered ----------------
__global__ __launch_bounds__(256) void gemm_fc_kernel(
    const _Float16* __restrict__ ah, const _Float16* __restrict__ wt,
    const float* __restrict__ bias, const float* __restrict__ xres,
    float* __restrict__ out) {
  __shared__ __align__(16) _Float16 As[2][128 * 40];
  __shared__ __align__(16) _Float16 Bs[2][128 * 40];
  const int tid = threadIdx.x, lane = tid & 31, wv = tid >> 5;
  const int m0 = blockIdx.y * 128, n0 = blockIdx.x * 128;
  const int wm = (wv & 3) * 32, wn = (wv >> 2) * 64;
  v8f acc[2][4] = {};

  const unsigned int asa0 = (unsigned int)(size_t)(void*)&As[0][0];
  const unsigned int asa1 = (unsigned int)(size_t)(void*)&As[1][0];
  const unsigned int bsa0 = (unsigned int)(size_t)(void*)&Bs[0][0];
  const unsigned int bsa1 = (unsigned int)(size_t)(void*)&Bs[1][0];
  const _Float16* abase = ah + (size_t)m0 * DIM;
  const _Float16* bbase = wt + (size_t)n0 * DIM;

  if (wv == 0) {
    tdm_load_tile_128x32(asa0, abase, MTOT);
    tdm_load_tile_128x32(bsa0, bbase, DIM);
  }

  int step = 0;
  for (int k0 = 0; k0 < DIM; k0 += 32, ++step) {
    if (wv == 0) {
      if (k0 + 32 < DIM) {
        unsigned int na = (step & 1) ? asa0 : asa1;
        unsigned int nb = (step & 1) ? bsa0 : bsa1;
        tdm_load_tile_128x32(na, abase + k0 + 32, MTOT);
        tdm_load_tile_128x32(nb, bbase + k0 + 32, DIM);
        WAIT_TENSORCNT(2);
      } else {
        WAIT_TENSORCNT(0);
      }
    }
    __syncthreads();
    const _Float16* a = As[step & 1];
    const _Float16* bb = Bs[step & 1];
    v16h a0 = load_frag(a + (wm + 0)  * 40, 40, lane);
    v16h a1 = load_frag(a + (wm + 16) * 40, 40, lane);
#pragma unroll
    for (int nt = 0; nt < 4; ++nt) {
      v16h bf = load_frag(bb + (wn + nt * 16) * 40, 40, lane);
      acc[0][nt] = wmma16(a0, bf, acc[0][nt]);
      acc[1][nt] = wmma16(a1, bf, acc[1][nt]);
    }
    __syncthreads();
  }

#pragma unroll
  for (int nt = 0; nt < 4; ++nt) {
    int gc = n0 + wn + nt * 16 + (lane & 15);
    float bv = bias[gc];
#pragma unroll
    for (int t = 0; t < 2; ++t)
#pragma unroll
      for (int r = 0; r < 8; ++r) {
        int gr = m0 + wm + t * 16 + r + ((lane >> 4) << 3);
        size_t o = (size_t)gr * DIM + gc;
        out[o] = acc[t][nt][r] + bv + xres[o];
      }
  }
}

extern "C" void kernel_launch(void* const* d_in, const int* in_sizes, int n_in,
                              void* d_out, int out_size, void* d_ws, size_t ws_size,
                              hipStream_t stream) {
  (void)in_sizes; (void)n_in; (void)out_size; (void)ws_size;
  const float* x     = (const float*)d_in[0];
  const float* w_qkv = (const float*)d_in[1];
  const float* b_qkv = (const float*)d_in[2];
  const float* w_fc  = (const float*)d_in[3];
  const float* b_fc  = (const float*)d_in[4];
  float* out = (float*)d_out;

  // workspace layout (all f16): ~86 MB total
  char* w = (char*)d_ws;
  _Float16* wqkvT = (_Float16*)w;  w += (size_t)NQKV * DIM * 2;
  _Float16* wfcT  = (_Float16*)w;  w += (size_t)DIM * DIM * 2;
  _Float16* xh    = (_Float16*)w;  w += (size_t)MTOT * DIM * 2;
  _Float16* qh    = (_Float16*)w;  w += (size_t)MTOT * DIM * 2;
  _Float16* kh    = (_Float16*)w;  w += (size_t)MTOT * DIM * 2;
  _Float16* vth   = (_Float16*)w;  w += (size_t)MTOT * DIM * 2;
  _Float16* ah    = (_Float16*)w;  w += (size_t)MTOT * DIM * 2;

  convert_x_kernel<<<((size_t)MTOT * DIM / 8 + 255) / 256, 256, 0, stream>>>(x, xh);
  transpose_w_kernel<<<(DIM * NQKV + 255) / 256, 256, 0, stream>>>(w_qkv, wqkvT, DIM, NQKV);
  transpose_w_kernel<<<(DIM * DIM + 255) / 256, 256, 0, stream>>>(w_fc, wfcT, DIM, DIM);
  gemm_qkv_kernel<<<dim3(NQKV / 128, MTOT / 128), 256, 0, stream>>>(xh, wqkvT, b_qkv, qh, kh, vth);
  attn_kernel<<<dim3(SEQ / 32, BATCH), 256, 0, stream>>>(qh, kh, vth, ah);
  gemm_fc_kernel<<<dim3(DIM / 128, MTOT / 128), 256, 0, stream>>>(ah, wfcT, b_fc, x, out);
}
